// EAST_85014582657132
// MI455X (gfx1250) — matrix operand order repaired and checked by reading, not compiled, not run on hardware
//
#include <hip/hip_runtime.h>
#include <hip/hip_bf16.h>

// ---------------------------------------------------------------------------
// ProbSparse attention (Informer) for MI455X / gfx1250.
// Heavy GEMMs (qkv projections, q@k^T) on v_wmma_f32_16x16x32_bf16 with a
// double-buffered global_load_async_to_lds_b128 pipeline (ASYNCcnt).
// ---------------------------------------------------------------------------

typedef __attribute__((ext_vector_type(16))) __bf16 v16bf;
typedef __attribute__((ext_vector_type(8)))  float v8f;
typedef int v4i __attribute__((vector_size(16)));   // matches builtin's V4i

#define Bb   8
#define Tt   12
#define Nn   500
#define Dd   512
#define K2   1024
#define Ss   35
#define BT   (Bb*Tt)      // 96
#define ROWS (Bb*Tt*Nn)   // 48000

// GEMM tiling: block tile 128x128, K-chunk 32, 8 waves in 4(M)x2(N),
// wave tile 32(M)x64(N) -> 8 WMMAs per wave per K-chunk.
#define TM   128
#define TN   128
#define TK   32
#define LSTR 40           // bf16 elems per LDS row (mult of 8 -> 16B aligned)

#if defined(__gfx1250__) && __has_builtin(__builtin_amdgcn_global_load_async_to_lds_b128)
#define HAS_ASYNC 1
#else
#define HAS_ASYNC 0
#endif

#if HAS_ASYNC
// builtin signature: void(v4i AS1* gsrc, v4i AS3* ldst, Ii offset, Ii cpol)
#define GPTR(x) ((__attribute__((address_space(1))) v4i*)(x))
#define LPTR(x) ((__attribute__((address_space(3))) v4i*)(x))
__device__ inline void wait_async0() {
#if __has_builtin(__builtin_amdgcn_s_wait_asynccnt)
  __builtin_amdgcn_s_wait_asynccnt(0);
#else
  asm volatile("s_wait_asynccnt 0x0" ::: "memory");
#endif
}
#endif

// ---- WMMA fragment loads from LDS (ISA 7.12.2 bf16 layouts) ----------------
// A 16x32: lane L (m=L%16, h=L/16) holds K = [8h..8h+7] and [16+8h..16+8h+7]
__device__ inline v16bf fragA(const __bf16* rowPtr, int h) {
  v16bf r;
#pragma unroll
  for (int i = 0; i < 8; ++i) {
    r[i]     = rowPtr[h * 8 + i];
    r[8 + i] = rowPtr[16 + h * 8 + i];
  }
  return r;
}
// B 32x16: lane L (n=L%16, h=L/16) holds K = [8h..8h+7] and [16+8h..16+8h+7]
__device__ inline v16bf fragB(const __bf16* rowPtr, int h) {
  v16bf r;
#pragma unroll
  for (int i = 0; i < 8; ++i) {
    r[i]     = rowPtr[h * 8 + i];
    r[8 + i] = rowPtr[16 + h * 8 + i];
  }
  return r;
}

#define WMMA_BF16(a, b, c) \
  __builtin_amdgcn_wmma_f32_16x16x32_bf16(false, (a), false, (b), (short)0, (c), false, false)

// ---------------------------------------------------------------------------
// Pre-pass 1: Xc = bf16(concat(X, STE))   [ROWS x 1024]
// ---------------------------------------------------------------------------
__global__ __launch_bounds__(256) void cvt_x_kernel(
    const float* __restrict__ X, const float* __restrict__ STE,
    __bf16* __restrict__ Xc)
{
  size_t i = (size_t)blockIdx.x * 256 + threadIdx.x;   // one float4 per source
  size_t row = i >> 7;          // 128 float4 per 512-wide row
  size_t c4  = i & 127;
  float4 x4 = ((const float4*)X)[i];
  float4 s4 = ((const float4*)STE)[i];
  __bf16* dx = Xc + row * K2 + c4 * 4;
  __bf16* ds = dx + Dd;
  dx[0] = (__bf16)x4.x; dx[1] = (__bf16)x4.y; dx[2] = (__bf16)x4.z; dx[3] = (__bf16)x4.w;
  ds[0] = (__bf16)s4.x; ds[1] = (__bf16)s4.y; ds[2] = (__bf16)s4.z; ds[3] = (__bf16)s4.w;
}

// ---------------------------------------------------------------------------
// Pre-pass 2: Wt[n][k] = bf16(W[k][n])   [512 x 1024] per matrix
// ---------------------------------------------------------------------------
__global__ __launch_bounds__(256) void cvt_w_kernel(
    const float* __restrict__ W, __bf16* __restrict__ Wt)
{
  int i = blockIdx.x * 256 + threadIdx.x;   // over K2*Dd, n fastest (coalesced read)
  int k = i >> 9, n = i & 511;
  Wt[(size_t)n * K2 + k] = (__bf16)W[i];
}

// ---------------------------------------------------------------------------
// Unified bf16 WMMA GEMM: C[row, col] = sum_k A[row,k] * Bt[col,k]
//   MODE 0: out = bf16(relu(C + bias[col]))          (q, k projections)
//   MODE 1: out = f32 (relu(C + bias[col]))          (v projection)
//   MODE 2: out = f32 C, row/col guarded to Nn       (attention scores)
// ---------------------------------------------------------------------------
template<int MODE>
__global__ __launch_bounds__(256) void gemm_bf16(
    const __bf16* __restrict__ Ag, const __bf16* __restrict__ Bg,
    const float* __restrict__ bias,
    float* __restrict__ outF, __bf16* __restrict__ outB,
    int K, int ldA, int ldB, int ldO,
    unsigned long long zsA, unsigned long long zsB, unsigned long long zsO)
{
  __shared__ __bf16 As[2][TM * LSTR];
  __shared__ __bf16 Bs[2][TN * LSTR];

  const __bf16* Ab = Ag + (size_t)blockIdx.z * zsA;
  const __bf16* Bt = Bg + (size_t)blockIdx.z * zsB;

  const int tid  = threadIdx.x;
  const int row0 = blockIdx.y * TM;
  const int col0 = blockIdx.x * TN;
  const int wave = tid >> 5, lane = tid & 31;
  const int wm = wave >> 1, wn = wave & 1;   // 4(M) x 2(N) waves
  const int h  = lane >> 4, ln = lane & 15;

  v8f acc[2][4] = {};
  const int nch = K / TK;

  // ---- tile copy: 128x32 bf16 A and 128x32 bf16 B per chunk ----
  auto issue = [&](int buf, int k0) {
#pragma unroll
    for (int e = tid; e < 512; e += 256) {       // A: 512 x 16B
      int r = e >> 2, c4 = e & 3;
      const __bf16* g = Ab + (size_t)(row0 + r) * ldA + k0 + c4 * 8;
      __bf16* l = &As[buf][r * LSTR + c4 * 8];
#if HAS_ASYNC
      __builtin_amdgcn_global_load_async_to_lds_b128(GPTR(g), LPTR(l), 0, 0);
#else
      *(uint4*)l = *(const uint4*)g;
#endif
    }
#pragma unroll
    for (int e = tid; e < 512; e += 256) {       // B: 512 x 16B
      int r = e >> 2, c4 = e & 3;
      const __bf16* g = Bt + (size_t)(col0 + r) * ldB + k0 + c4 * 8;
      __bf16* l = &Bs[buf][r * LSTR + c4 * 8];
#if HAS_ASYNC
      __builtin_amdgcn_global_load_async_to_lds_b128(GPTR(g), LPTR(l), 0, 0);
#else
      *(uint4*)l = *(const uint4*)g;
#endif
    }
  };

  auto compute = [&](int cur) {
    v16bf a0 = fragA(&As[cur][(wm * 32 +      ln) * LSTR], h);
    v16bf a1 = fragA(&As[cur][(wm * 32 + 16 + ln) * LSTR], h);
    v16bf b0 = fragB(&Bs[cur][(wn * 64 +      ln) * LSTR], h);
    v16bf b1 = fragB(&Bs[cur][(wn * 64 + 16 + ln) * LSTR], h);
    v16bf b2 = fragB(&Bs[cur][(wn * 64 + 32 + ln) * LSTR], h);
    v16bf b3 = fragB(&Bs[cur][(wn * 64 + 48 + ln) * LSTR], h);
    acc[0][0] = WMMA_BF16(a0, b0, acc[0][0]);
    acc[0][1] = WMMA_BF16(a0, b1, acc[0][1]);
    acc[0][2] = WMMA_BF16(a0, b2, acc[0][2]);
    acc[0][3] = WMMA_BF16(a0, b3, acc[0][3]);
    acc[1][0] = WMMA_BF16(a1, b0, acc[1][0]);
    acc[1][1] = WMMA_BF16(a1, b1, acc[1][1]);
    acc[1][2] = WMMA_BF16(a1, b2, acc[1][2]);
    acc[1][3] = WMMA_BF16(a1, b3, acc[1][3]);
  };

#if HAS_ASYNC
  // double-buffered async pipeline: prefetch next chunk during WMMA
  issue(0, 0);
  wait_async0();
  __syncthreads();
  for (int kc = 0; kc < nch; ++kc) {
    int cur = kc & 1;
    if (kc + 1 < nch) issue(cur ^ 1, (kc + 1) * TK);
    compute(cur);
    wait_async0();
    __syncthreads();
  }
#else
  for (int kc = 0; kc < nch; ++kc) {
    __syncthreads();
    issue(0, kc * TK);
    __syncthreads();
    compute(0);
  }
#endif

  // ---- epilogue ----
#pragma unroll
  for (int i = 0; i < 2; ++i)
#pragma unroll
    for (int j = 0; j < 4; ++j) {
      int col = col0 + wn * 64 + j * 16 + ln;
      float bv = (MODE == 2) ? 0.f : bias[col];
#pragma unroll
      for (int r = 0; r < 8; ++r) {
        int row = row0 + wm * 32 + i * 16 + h * 8 + r;
        float val = acc[i][j][r];
        if (MODE != 2) {
          val += bv;
          val = val > 0.f ? val : 0.f;
        }
        if (MODE == 0) {
          outB[(size_t)row * ldO + col] = (__bf16)val;
        } else if (MODE == 1) {
          outF[(size_t)row * ldO + col] = val;
        } else {
          if (row < Nn && col < Nn)
            outF[(size_t)blockIdx.z * zsO + (size_t)row * ldO + col] = val;
        }
      }
    }
}

// ---------------------------------------------------------------------------
// M[l] = max_s scores[l, samp[l,s]] - (sum_s ...) / N
// ---------------------------------------------------------------------------
__global__ __launch_bounds__(512) void mscore_kernel(
    const float* __restrict__ scores, const int* __restrict__ samp,
    float* __restrict__ Mbuf)
{
  int bt = blockIdx.x, l = threadIdx.x;
  if (l >= Nn) return;
  const float* srow = scores + ((size_t)bt * Nn + l) * Nn;
  float mx = -INFINITY, sm = 0.f;
#pragma unroll 5
  for (int s = 0; s < Ss; ++s) {
    float v = srow[samp[l * Ss + s]];
    mx = fmaxf(mx, v);
    sm += v;
  }
  Mbuf[bt * Nn + l] = mx - sm * (1.f / (float)Nn);
}

// ---------------------------------------------------------------------------
// top-35 of M per (b,t); jax tie-break = lowest index
// ---------------------------------------------------------------------------
__global__ __launch_bounds__(512) void topk_kernel(
    const float* __restrict__ Mbuf, int* __restrict__ idxbuf)
{
  __shared__ float vals[512];
  __shared__ float rv[512];
  __shared__ int   ri[512];
  int bt = blockIdx.x, t = threadIdx.x;
  vals[t] = (t < Nn) ? Mbuf[bt * Nn + t] : -INFINITY;
  __syncthreads();
  for (int it = 0; it < Ss; ++it) {
    rv[t] = vals[t]; ri[t] = t;
    __syncthreads();
    for (int off = 256; off > 0; off >>= 1) {
      if (t < off) {
        float v2 = rv[t + off]; int i2 = ri[t + off];
        if (v2 > rv[t] || (v2 == rv[t] && i2 < ri[t])) { rv[t] = v2; ri[t] = i2; }
      }
      __syncthreads();
    }
    if (t == 0) { idxbuf[bt * Ss + it] = ri[0]; vals[ri[0]] = -INFINITY; }
    __syncthreads();
  }
}

// ---------------------------------------------------------------------------
// causal-masked scaled softmax over selected row + attn @ v
// ---------------------------------------------------------------------------
__global__ __launch_bounds__(512) void attn_kernel(
    const float* __restrict__ scores, const float* __restrict__ vf,
    const int* __restrict__ idxbuf, float* __restrict__ ctxnew)
{
  __shared__ float sp[512];
  __shared__ float red[512];
  int u = blockIdx.x, bt = blockIdx.y, t = threadIdx.x;
  int r = idxbuf[bt * Ss + u];
  const float scale = 0.044194173824159216f;  // 1/sqrt(512)
  const float* srow = scores + ((size_t)bt * Nn + r) * Nn;

  float lv = (t <= r) ? srow[t] * scale : -INFINITY;
  red[t] = lv;
  __syncthreads();
  for (int off = 256; off > 0; off >>= 1) {
    if (t < off) red[t] = fmaxf(red[t], red[t + off]);
    __syncthreads();
  }
  float mx = red[0];
  __syncthreads();

  float e = (t <= r) ? __expf(lv - mx) : 0.f;
  sp[t] = e; red[t] = e;
  __syncthreads();
  for (int off = 256; off > 0; off >>= 1) {
    if (t < off) red[t] += red[t + off];
    __syncthreads();
  }
  float inv = 1.f / red[0];

  const float* vb = vf + (size_t)bt * Nn * Dd;
  float acc = 0.f;
  for (int m = 0; m <= r; ++m) acc += sp[m] * vb[(size_t)m * Dd + t];
  ctxnew[((size_t)bt * Ss + u) * Dd + t] = acc * inv;
}

// ---------------------------------------------------------------------------
// context = cumsum(v, axis=N), written in output layout [B,N,T,D]
// ---------------------------------------------------------------------------
__global__ __launch_bounds__(512) void cumsum_kernel(
    const float* __restrict__ vf, float* __restrict__ out)
{
  int bt = blockIdx.x, d = threadIdx.x;
  int b = bt / Tt, tt = bt % Tt;
  const float* vb = vf + (size_t)bt * Nn * Dd;
  float acc = 0.f;
  for (int n = 0; n < Nn; ++n) {
    acc += vb[(size_t)n * Dd + d];
    out[(((size_t)b * Nn + n) * Tt + tt) * Dd + d] = acc;
  }
}

// ---------------------------------------------------------------------------
// scatter ctx_new rows into output
// ---------------------------------------------------------------------------
__global__ __launch_bounds__(512) void scatter_kernel(
    const float* __restrict__ ctxnew, const int* __restrict__ idxbuf,
    float* __restrict__ out)
{
  int u = blockIdx.x, bt = blockIdx.y, d = threadIdx.x;
  int b = bt / Tt, tt = bt % Tt;
  int r = idxbuf[bt * Ss + u];
  out[(((size_t)b * Nn + r) * Tt + tt) * Dd + d] =
      ctxnew[((size_t)bt * Ss + u) * Dd + d];
}

// ---------------------------------------------------------------------------
extern "C" void kernel_launch(void* const* d_in, const int* in_sizes, int n_in,
                              void* d_out, int out_size, void* d_ws, size_t ws_size,
                              hipStream_t stream)
{
  const float* X    = (const float*)d_in[0];
  const float* STE  = (const float*)d_in[1];
  const float* Wq   = (const float*)d_in[2];
  const float* bq   = (const float*)d_in[3];
  const float* Wk   = (const float*)d_in[4];
  const float* bk   = (const float*)d_in[5];
  const float* Wv   = (const float*)d_in[6];
  const float* bv   = (const float*)d_in[7];
  const int*   samp = (const int*)d_in[8];
  float* out = (float*)d_out;

  // workspace carving (256B aligned)
  char* ws = (char*)d_ws;
  size_t off = 0;
  auto carve = [&](size_t bytes) {
    size_t o = off;
    off = (off + bytes + 255) & ~(size_t)255;
    return o;
  };
  __bf16* Xc     = (__bf16*)(ws + carve((size_t)ROWS * K2 * sizeof(__bf16)));
  __bf16* Wtq    = (__bf16*)(ws + carve((size_t)Dd * K2 * sizeof(__bf16)));
  __bf16* Wtk    = (__bf16*)(ws + carve((size_t)Dd * K2 * sizeof(__bf16)));
  __bf16* Wtv    = (__bf16*)(ws + carve((size_t)Dd * K2 * sizeof(__bf16)));
  __bf16* qb     = (__bf16*)(ws + carve((size_t)ROWS * Dd * sizeof(__bf16)));
  __bf16* kb     = (__bf16*)(ws + carve((size_t)ROWS * Dd * sizeof(__bf16)));
  float*  vf     = (float*) (ws + carve((size_t)ROWS * Dd * sizeof(float)));
  float*  scores = (float*) (ws + carve((size_t)BT * Nn * Nn * sizeof(float)));
  float*  Mbuf   = (float*) (ws + carve((size_t)BT * Nn * sizeof(float)));
  int*    idxbuf = (int*)   (ws + carve((size_t)BT * Ss * sizeof(int)));
  float*  ctxnew = (float*) (ws + carve((size_t)BT * Ss * Dd * sizeof(float)));

  // pre-pass: bf16 activations + transposed bf16 weights
  cvt_x_kernel<<<(ROWS * (Dd / 4)) / 256, 256, 0, stream>>>(X, STE, Xc);
  cvt_w_kernel<<<(K2 * Dd) / 256, 256, 0, stream>>>(Wq, Wtq);
  cvt_w_kernel<<<(K2 * Dd) / 256, 256, 0, stream>>>(Wk, Wtk);
  cvt_w_kernel<<<(K2 * Dd) / 256, 256, 0, stream>>>(Wv, Wtv);

  // qkv projections: [48000 x 1024] x [1024 x 512]
  dim3 gProj(Dd / TN, ROWS / TM);           // (4, 375)
  gemm_bf16<0><<<gProj, 256, 0, stream>>>(Xc, Wtq, bq, nullptr, qb,
                                          K2, K2, K2, Dd, 0ull, 0ull, 0ull);
  gemm_bf16<0><<<gProj, 256, 0, stream>>>(Xc, Wtk, bk, nullptr, kb,
                                          K2, K2, K2, Dd, 0ull, 0ull, 0ull);
  gemm_bf16<1><<<gProj, 256, 0, stream>>>(Xc, Wtv, bv, vf, nullptr,
                                          K2, K2, K2, Dd, 0ull, 0ull, 0ull);

  // scores_full[bt] = q[bt] @ k[bt]^T : [500 x 512] x [512 x 500] per (b,t)
  dim3 gSc((Nn + TN - 1) / TN, (Nn + TM - 1) / TM, BT);   // (4, 4, 96)
  gemm_bf16<2><<<gSc, 256, 0, stream>>>(qb, kb, nullptr, scores, nullptr,
                                        Dd, Dd, Dd, Nn,
                                        (unsigned long long)(Nn * Dd),
                                        (unsigned long long)(Nn * Dd),
                                        (unsigned long long)(Nn * Nn));

  mscore_kernel<<<BT, 512, 0, stream>>>(scores, samp, Mbuf);
  topk_kernel<<<BT, 512, 0, stream>>>(Mbuf, idxbuf);

  dim3 gAt(Ss, BT);                          // (35, 96)
  attn_kernel<<<gAt, 512, 0, stream>>>(scores, vf, idxbuf, ctxnew);

  cumsum_kernel<<<BT, 512, 0, stream>>>(vf, out);
  scatter_kernel<<<gAt, 512, 0, stream>>>(ctxnew, idxbuf, out);
}
